// GraphEncoder_16338055594643
// MI455X (gfx1250) — matrix operand order repaired and checked by reading, not compiled, not run on hardware
//
#include <hip/hip_runtime.h>

// ---------------------------------------------------------------------------
// Types for CDNA5 WMMA (wave32): v_wmma_f32_16x16x32_bf16
// ---------------------------------------------------------------------------
typedef __attribute__((ext_vector_type(16))) __bf16 v16bf;
typedef __attribute__((ext_vector_type(8)))  float  v8f;

union FragBF {           // 16 bf16 = 32 bytes, assembled from two 16B LDS reads
  uint4 q[2];
  v16bf v;
};

__device__ __forceinline__ unsigned short f2bf(float f) {
  union { float f; unsigned u; } x;
  x.f = f;
  unsigned u = x.u + (0x7FFFu + ((x.u >> 16) & 1u));   // round-to-nearest-even
  return (unsigned short)(u >> 16);
}

__device__ __forceinline__ unsigned pk2bf(float lo, float hi) {
  return (unsigned)f2bf(lo) | ((unsigned)f2bf(hi) << 16);
}

// ---------------------------------------------------------------------------
// Tiled WMMA GEMM:  C[M,N] = act(A[M,K] @ B[K,N] + bias[N])
// 256 threads = 8 waves; block tile 64x64, K stepped by 32.
// Wave grid 4x2; each wave owns a 16x32 C strip = two 16x16 WMMA tiles,
// reusing its A fragment across both -> 2 WMMAs per barrier pair.
// Staging: A rows clamped to M-1 and B cols clamped to N-1 (out-of-range
// rows/cols only feed C entries that are never stored), so the main K loop
// is completely branch/select-free; only the K remainder needs zero-fill.
// ---------------------------------------------------------------------------
template <bool RELU>
__global__ __launch_bounds__(256) void gemm_bias_act(
    const float* __restrict__ A, const float* __restrict__ B,
    const float* __restrict__ bias, float* __restrict__ C,
    int M, int N, int K)
{
  __shared__ __align__(16) unsigned short As[64][32];   // [m][k] bf16
  __shared__ __align__(16) unsigned short Bs[64][32];   // [n][k] bf16 (B transposed)

  const int t    = threadIdx.x;
  const int wave = t >> 5;
  const int lane = t & 31;
  const int bm0  = blockIdx.x * 64;
  const int bn0  = blockIdx.y * 64;

  const int wm  = (wave >> 1) * 16;   // 0,16,32,48
  const int wn0 = (wave & 1)  * 32;   // 0,32  (two 16-wide tiles: wn0, wn0+16)
  const int grp = lane >> 4;          // lane half
  const int l16 = lane & 15;

  // ---- staging assignments ----
  // A: thread t loads 8 contiguous floats of one A row: row = t/4, col = (t%4)*8
  const int a_row = t >> 2;
  const int a_col = (t & 3) * 8;
  const int gm    = min(bm0 + a_row, M - 1);            // clamp, never zero-fill
  const float* __restrict__ Arow = A + (size_t)gm * K;

  // B: thread t loads 8 contiguous floats of one B row: k = t/8, n = (t%8)*8,
  // then transposes into Bs[n][k] via 8 b16 LDS stores.
  const int b_kk = t >> 3;            // 0..31  (K offset within chunk)
  const int b_nb = (t & 7) * 8;       // 0..56  (N offset within tile)

  v8f acc0 = {};
  v8f acc1 = {};

  // ---- one barrier pair, two WMMAs (A fragment reused) ----
  auto compute = [&]() {
    __syncthreads();
    FragBF fa, fb0, fb1;
    // A 16x32: lane group 0 -> K 0..7 then 16..23 ; group 1 -> K 8..15, 24..31
    fa.q[0]  = *reinterpret_cast<const uint4*>(&As[wm + l16][grp * 8]);
    fa.q[1]  = *reinterpret_cast<const uint4*>(&As[wm + l16][16 + grp * 8]);
    // B 32x16: lane group 0 -> K 0..7 then 16..23 ; group 1 -> K 8..15, 24..31 (column n = lane%16)
    fb0.q[0] = *reinterpret_cast<const uint4*>(&Bs[wn0 + l16][grp * 8]);
    fb0.q[1] = *reinterpret_cast<const uint4*>(&Bs[wn0 + l16][16 + grp * 8]);
    fb1.q[0] = *reinterpret_cast<const uint4*>(&Bs[wn0 + 16 + l16][grp * 8]);
    fb1.q[1] = *reinterpret_cast<const uint4*>(&Bs[wn0 + 16 + l16][16 + grp * 8]);
    acc0 = __builtin_amdgcn_wmma_f32_16x16x32_bf16(
        false, fa.v, false, fb0.v, (short)0, acc0, false, false);
    acc1 = __builtin_amdgcn_wmma_f32_16x16x32_bf16(
        false, fa.v, false, fb1.v, (short)0, acc1, false, false);
    __syncthreads();
  };

  // ---- fast staging: unconditional b128 loads, no clamps/selects ----
  auto stage_fast = [&](int k0) {
    const float4 v0 = *reinterpret_cast<const float4*>(Arow + k0 + a_col);
    const float4 v1 = *reinterpret_cast<const float4*>(Arow + k0 + a_col + 4);
    uint4 q;
    q.x = pk2bf(v0.x, v0.y); q.y = pk2bf(v0.z, v0.w);
    q.z = pk2bf(v1.x, v1.y); q.w = pk2bf(v1.z, v1.w);
    *reinterpret_cast<uint4*>(&As[a_row][a_col]) = q;

    const float* __restrict__ Brow = B + (size_t)(k0 + b_kk) * N + bn0 + b_nb;
    const float4 w0 = *reinterpret_cast<const float4*>(Brow);
    const float4 w1 = *reinterpret_cast<const float4*>(Brow + 4);
    Bs[b_nb + 0][b_kk] = f2bf(w0.x);
    Bs[b_nb + 1][b_kk] = f2bf(w0.y);
    Bs[b_nb + 2][b_kk] = f2bf(w0.z);
    Bs[b_nb + 3][b_kk] = f2bf(w0.w);
    Bs[b_nb + 4][b_kk] = f2bf(w1.x);
    Bs[b_nb + 5][b_kk] = f2bf(w1.y);
    Bs[b_nb + 6][b_kk] = f2bf(w1.z);
    Bs[b_nb + 7][b_kk] = f2bf(w1.w);
  };

  // ---- generic staging: clamped addresses + branchless zero at K tail ----
  auto stage_slow = [&](int k0) {
    float a[8];
#pragma unroll
    for (int j = 0; j < 8; ++j) {
      const int gk = k0 + a_col + j;
      const float v = Arow[min(gk, K - 1)];
      a[j] = (gk < K) ? v : 0.0f;
    }
    uint4 q;
    q.x = pk2bf(a[0], a[1]); q.y = pk2bf(a[2], a[3]);
    q.z = pk2bf(a[4], a[5]); q.w = pk2bf(a[6], a[7]);
    *reinterpret_cast<uint4*>(&As[a_row][a_col]) = q;

    const int gk  = k0 + b_kk;
    const int gkc = min(gk, K - 1);
#pragma unroll
    for (int j = 0; j < 8; ++j) {
      const float v = B[(size_t)gkc * N + min(bn0 + b_nb + j, N - 1)];
      Bs[b_nb + j][b_kk] = f2bf((gk < K) ? v : 0.0f);
    }
  };

  const int  kfull  = K & ~31;
  const bool fastOK = ((K & 3) == 0) && ((N & 3) == 0) && (bn0 + 64 <= N);

  if (fastOK) {
    for (int k0 = 0; k0 < kfull; k0 += 32) { stage_fast(k0); compute(); }
  } else {
    for (int k0 = 0; k0 < kfull; k0 += 32) { stage_slow(k0); compute(); }
  }
  if (kfull < K) { stage_slow(kfull); compute(); }

  // ---- epilogue: bias + optional ReLU; C/D layout: lane%16 = col,
  // row = wm + 8*(lane/16) + r. Stores coalesce across lanes. ----
#pragma unroll
  for (int tile = 0; tile < 2; ++tile) {
    const int col = bn0 + wn0 + tile * 16 + l16;
    if (col < N) {
      const float bv = bias ? bias[col] : 0.0f;
      const v8f& acc = tile ? acc1 : acc0;
#pragma unroll
      for (int r = 0; r < 8; ++r) {
        const int row = bm0 + wm + grp * 8 + r;
        if (row < M) {
          float v = acc[r] + bv;
          if (RELU) v = fmaxf(v, 0.0f);
          C[(size_t)row * N + col] = v;
        }
      }
    }
  }
}

// ---------------------------------------------------------------------------
// One block per node: aggr[n,f] = h[n,f] + emb1[0,f] + emb2[0,f]
// (self-loop with zero edge_attr; fully initializes aggr)
// ---------------------------------------------------------------------------
__global__ void self_loop_init_kernel(const float* __restrict__ h,
                                      const float* __restrict__ emb1,
                                      const float* __restrict__ emb2,
                                      float* __restrict__ aggr, int F)
{
  const int n = blockIdx.x;
  const float* __restrict__ hn = h + (size_t)n * F;
  float* __restrict__ an = aggr + (size_t)n * F;
  for (int f = threadIdx.x; f < F; f += blockDim.x)
    an[f] = hn[f] + emb1[f] + emb2[f];
}

// ---------------------------------------------------------------------------
// One block per edge: aggr[dst,:] += h[src,:] + emb1[t0,:] + emb2[t1,:]
// ---------------------------------------------------------------------------
__global__ void edge_scatter_kernel(const float* __restrict__ h,
                                    const int* __restrict__ edge_index, // [2,E]
                                    const int* __restrict__ edge_attr,  // [E,2]
                                    const float* __restrict__ emb1,
                                    const float* __restrict__ emb2,
                                    float* __restrict__ aggr,
                                    int E, int F)
{
  const int e  = blockIdx.x;
  const int s  = edge_index[e];
  const int d  = edge_index[E + e];
  const int t0 = edge_attr[2 * e];
  const int t1 = edge_attr[2 * e + 1];
  const float* __restrict__ hs = h    + (size_t)s  * F;
  const float* __restrict__ e1 = emb1 + (size_t)t0 * F;
  const float* __restrict__ e2 = emb2 + (size_t)t1 * F;
  float* __restrict__ ad = aggr + (size_t)d * F;
  for (int f = threadIdx.x; f < F; f += blockDim.x)
    unsafeAtomicAdd(&ad[f], hs[f] + e1[f] + e2[f]);
}

// ---------------------------------------------------------------------------
// One block per node: g[batch[n], :] += h[n, :]
// ---------------------------------------------------------------------------
__global__ void pool_scatter_kernel(const float* __restrict__ h,
                                    const int* __restrict__ batch,
                                    float* __restrict__ g, int F)
{
  const int n = blockIdx.x;
  const float* __restrict__ hn = h + (size_t)n * F;
  float* __restrict__ gd = g + (size_t)batch[n] * F;
  for (int f = threadIdx.x; f < F; f += blockDim.x)
    unsafeAtomicAdd(&gd[f], hn[f]);
}

__global__ void zero_kernel(float* __restrict__ p, int n)
{
  const int i = blockIdx.x * blockDim.x + threadIdx.x;
  if (i < n) p[i] = 0.0f;
}

// ---------------------------------------------------------------------------
// Launch
// ---------------------------------------------------------------------------
static inline int cdiv(int a, int b) { return (a + b - 1) / b; }

extern "C" void kernel_launch(void* const* d_in, const int* in_sizes, int n_in,
                              void* d_out, int out_size, void* d_ws, size_t ws_size,
                              hipStream_t stream)
{
  const float* x    = (const float*)d_in[0];
  const int*   ei   = (const int*)  d_in[1];
  const int*   ea   = (const int*)  d_in[2];
  const int*   bat  = (const int*)  d_in[3];
  const float* emb1 = (const float*)d_in[4];
  const float* emb2 = (const float*)d_in[5];
  const float* W1   = (const float*)d_in[6];
  const float* b1   = (const float*)d_in[7];
  const float* W2   = (const float*)d_in[8];
  const float* b2   = (const float*)d_in[9];
  const float* Wfc  = (const float*)d_in[10];
  const float* bfc  = (const float*)d_in[11];

  const int F  = in_sizes[11];          // HIDDEN == N_FEAT == 300
  const int H2 = in_sizes[7];           // 2*F == 600
  const int Nn = in_sizes[3];           // 100000 nodes (batch array length)
  const int E  = in_sizes[1] / 2;       // 200000 edges
  const int G  = out_size / F;          // 2048 graphs

  // Workspace layout (floats)
  float* A0 = (float*)d_ws;                 // aggr   [Nn, F]
  float* A1 = A0 + (size_t)Nn * F;          // mid    [Nn, 2F]
  float* A2 = A1 + (size_t)Nn * H2;         // hidden [Nn, F]
  float* g  = A2 + (size_t)Nn * F;          // pooled [G, F]

  int ftb = cdiv(F, 32) * 32;               // threads covering one feature row
  if (ftb > 1024) ftb = 1024;

  const int gf_total = G * F;
  dim3 gemm_blk(256);

  // ---------------- Layer 1: gin(x) ----------------
  self_loop_init_kernel<<<Nn, ftb, 0, stream>>>(x, emb1, emb2, A0, F);
  edge_scatter_kernel<<<E, ftb, 0, stream>>>(x, ei, ea, emb1, emb2, A0, E, F);
  // mid = relu(aggr @ W1 + b1)   [Nn, 2F]
  gemm_bias_act<true><<<dim3(cdiv(Nn, 64), cdiv(H2, 64)), gemm_blk, 0, stream>>>(A0, W1, b1, A1, Nn, H2, F);
  // h = relu(mid @ W2 + b2)      [Nn, F]   (outer relu fused: h = relu(gin(x)))
  gemm_bias_act<true><<<dim3(cdiv(Nn, 64), cdiv(F, 64)), gemm_blk, 0, stream>>>(A1, W2, b2, A2, Nn, F, H2);

  // ---------------- Layer 2: gin(h) ----------------
  self_loop_init_kernel<<<Nn, ftb, 0, stream>>>(A2, emb1, emb2, A0, F);
  edge_scatter_kernel<<<E, ftb, 0, stream>>>(A2, ei, ea, emb1, emb2, A0, E, F);
  gemm_bias_act<true><<<dim3(cdiv(Nn, 64), cdiv(H2, 64)), gemm_blk, 0, stream>>>(A0, W1, b1, A1, Nn, H2, F);
  gemm_bias_act<true><<<dim3(cdiv(Nn, 64), cdiv(F, 64)), gemm_blk, 0, stream>>>(A1, W2, b2, A2, Nn, F, H2);

  // ---------------- Global add pool + final linear ----------------
  zero_kernel<<<cdiv(gf_total, 256), 256, 0, stream>>>(g, gf_total);
  pool_scatter_kernel<<<Nn, ftb, 0, stream>>>(A2, bat, g, F);
  gemm_bias_act<false><<<dim3(cdiv(G, 64), cdiv(F, 64)), gemm_blk, 0, stream>>>(g, Wfc, bfc, (float*)d_out, G, F, F);
}
